// GCN_Two_GSO_11493332484239
// MI455X (gfx1250) — matrix operand order repaired and checked by reading, not compiled, not run on hardware
//
#include <hip/hip_runtime.h>
#include <hip/hip_bf16.h>

// ---------------------------------------------------------------------------
// GCN with two GSOs on MI455X (gfx1250).
//   deg -> (w_self[N], w_edge[E]) -> 3x { WMMA bf16 GEMM, init, scatter-add }
// Feature matrices (51.2MB) fit in the 192MB L2, so the random edge gathers
// and f32 atomics resolve in L2. GEMMs use v_wmma_f32_16x16x32_bf16 with the
// weight matrix staged in LDS (pre-swizzled to the B-operand lane layout).
// MSHIFT/RELU are template params so all store offsets are immediates.
// ---------------------------------------------------------------------------

typedef __attribute__((ext_vector_type(16))) __bf16 v16bf;
typedef __attribute__((ext_vector_type(8)))  float  v8f;

// pow with the reference's inf -> 0 masking (0 ** negative exponent)
__device__ __forceinline__ float pow0(float b, float e) {
    float p = powf(b, e);
    return isinf(p) ? 0.0f : p;
}

__global__ void k_zero(float* __restrict__ p, int n) {
    int i = blockIdx.x * blockDim.x + threadIdx.x;
    if (i < n) p[i] = 0.0f;
}

__global__ void k_degree(const int* __restrict__ row, float* __restrict__ deg, int E) {
    int e = blockIdx.x * blockDim.x + threadIdx.x;
    if (e < E) atomicAdd(deg + row[e], 1.0f);
}

// params layout: [m1, m2, m3, e1, e2, e3, a]
__global__ void k_wself(const float* __restrict__ deg,
                        const float* __restrict__ p1, const float* __restrict__ p2,
                        float* __restrict__ wself, int N) {
    int i = blockIdx.x * blockDim.x + threadIdx.x;
    if (i >= N) return;
    float dg = deg[i];
    float d1 = dg + p1[6];
    float d2 = dg + p2[6];
    float s = p1[0] * pow0(d1, p1[3]) + p1[1] * pow0(d1, p1[4]) * pow0(d1, p1[5]) + p1[2]
            + p2[0] * pow0(d2, p2[3]) + p2[1] * pow0(d2, p2[4]) * pow0(d2, p2[5]) + p2[2];
    wself[i] = s;
}

__global__ void k_wedge(const float* __restrict__ deg,
                        const int* __restrict__ row, const int* __restrict__ col,
                        const float* __restrict__ p1, const float* __restrict__ p2,
                        float* __restrict__ wedge, int E) {
    int e = blockIdx.x * blockDim.x + threadIdx.x;
    if (e >= E) return;
    float dr = deg[row[e]];
    float dc = deg[col[e]];
    float w = p1[1] * pow0(dr + p1[6], p1[4]) * pow0(dc + p1[6], p1[5])
            + p2[1] * pow0(dr + p2[6], p2[4]) * pow0(dc + p2[6], p2[5]);
    wedge[e] = w;
}

// ---------------------------------------------------------------------------
// T[nrows x MCOLS] = (RELU?) A[nrows x 128] @ W[128 x MCOLS], bf16 WMMA, f32 acc.
// One wave per 16-row tile; 8 waves / 128 rows per block. MCOLS = 1<<MSHIFT.
//
// W staged in LDS once per block, pre-swizzled so that for fragment
// frag = nt*4 + kc, lane "lane" holds its 16 B-elements contiguously:
//   Wl[frag*768 + lane*24 + i]  (24-elem / 48B stride keeps 16B alignment
//   for the two ds_load_b128 and reduces bank conflicts to 2-way).
// B element mapping (ISA 7.12.2, 16-bit 32x16): lane = h*16 + l, n = nt*16+l,
//   element i<8 -> K = kc*32 + h*8 + i ; i>=8 -> K = kc*32+16+h*8+(i-8).
// A layout (16-bit 16x32): l = lane&15 is M; h = lane>>4;
//   element i<8 -> K = kc*32 + h*8 + i ; element i>=8 -> K = kc*32+16+h*8+(i-8)
// D layout: VGPR r -> (M = h*8 + r, N = nt*16 + l)
// ---------------------------------------------------------------------------
template <int MSHIFT, bool RELU>
__global__ __launch_bounds__(256) void k_gemm_wmma(
    const float* __restrict__ A, const float* __restrict__ Wm,
    float* __restrict__ T, int nrows) {
    constexpr int K      = 128;
    constexpr int MCOLS  = 1 << MSHIFT;
    constexpr int MTILES = MCOLS >> 4;
    __shared__ __bf16 Wl[MTILES * 4 * 768];   // fragments * 32 lanes * 24 elems

    // ---- cooperative stage of W (coalesced f32 read -> swizzled bf16 LDS) ----
    constexpr int KELEMS = K * MCOLS;         // multiple of 256
#pragma unroll 4
    for (int idx = threadIdx.x; idx < KELEMS; idx += 256) {
        int k  = idx >> MSHIFT;
        int n  = idx & (MCOLS - 1);
        int kc = k >> 5;
        int kr = k & 31;
        int lane = ((kr & 8) << 1) + (n & 15); // h*16 + l
        int i    = (kr & 7) | ((kr >> 1) & 8);
        int frag = ((n >> 4) << 2) + kc;      // nt*4 + kc
        Wl[frag * 768 + lane * 24 + i] = (__bf16)Wm[idx];
    }
    __syncthreads();

    int wave = blockIdx.x * (blockDim.x >> 5) + (threadIdx.x >> 5);
    int row0 = wave << 4;
    if (row0 >= nrows) return;                // wave-uniform
    int lane = threadIdx.x & 31;
    int h = lane >> 4;
    int l = lane & 15;
    int arow = row0 + l;

    // ---- load + convert A once per wave: 4 K-chunks of 32, float4 loads ----
    v16bf a[4];
#pragma unroll
    for (int kc = 0; kc < 4; ++kc) {
        const float4* ap = (const float4*)(A + (size_t)arow * K + kc * 32 + h * 8);
        float4 x0 = ap[0], x1 = ap[1];        // K = +0..7
        float4 y0 = ap[4], y1 = ap[5];        // K = +16..23
        float xs[8] = {x0.x, x0.y, x0.z, x0.w, x1.x, x1.y, x1.z, x1.w};
        float ys[8] = {y0.x, y0.y, y0.z, y0.w, y1.x, y1.y, y1.z, y1.w};
#pragma unroll
        for (int i = 0; i < 8; ++i) {
            float xv = RELU ? fmaxf(xs[i], 0.0f) : xs[i];
            float yv = RELU ? fmaxf(ys[i], 0.0f) : ys[i];
            a[kc][i]     = (__bf16)xv;
            a[kc][8 + i] = (__bf16)yv;
        }
    }

#pragma unroll 2
    for (int nt = 0; nt < MTILES; ++nt) {     // wave-uniform, constant bound
        int n = (nt << 4) + l;
        v8f acc = {};
#pragma unroll
        for (int kc = 0; kc < 4; ++kc) {
            int frag = (nt << 2) + kc;
            v16bf b = *(const v16bf*)(Wl + frag * 768 + lane * 24);  // 2x ds_load_b128
            acc = __builtin_amdgcn_wmma_f32_16x16x32_bf16(
                false, a[kc], false, b, (short)0, acc, false, false);
        }
        // one base pointer + compile-time row offsets -> immediate-offset stores
        float* tp = T + ((size_t)(row0 + h * 8) << MSHIFT) + n;
#pragma unroll
        for (int r = 0; r < 8; ++r)
            tp[r << MSHIFT] = acc[r];
    }
}

// out[i,d] = w_self[i] * T[i,d] + bias[d]        (D = 1<<dshift)
__global__ void k_init(const float* __restrict__ T, const float* __restrict__ wself,
                       const float* __restrict__ bias, float* __restrict__ out,
                       int total, int dshift) {
    int idx = blockIdx.x * blockDim.x + threadIdx.x;
    if (idx >= total) return;
    int i = idx >> dshift;
    int d = idx & ((1 << dshift) - 1);
    out[idx] = wself[i] * T[idx] + bias[d];
}

// out[row[e]] += w_edge[e] * T[col[e]]  -- one float4 group per thread, f32 atomics.
// g4shift = log2(D/4): 5 for D=128, 4 for D=64.
__global__ void k_scatter(const float* __restrict__ T, const float* __restrict__ wedge,
                          const int* __restrict__ row, const int* __restrict__ col,
                          float* __restrict__ out, int total, int g4shift, int D) {
    int idx = blockIdx.x * blockDim.x + threadIdx.x;
    if (idx >= total) return;
    int e = idx >> g4shift;
    int c = (idx & ((1 << g4shift) - 1)) << 2;
    float w = wedge[e];
    const float4 v = *(const float4*)(T + (size_t)col[e] * D + c);
    float* dst = out + (size_t)row[e] * D + c;
    atomicAdd(dst + 0, w * v.x);
    atomicAdd(dst + 1, w * v.y);
    atomicAdd(dst + 2, w * v.z);
    atomicAdd(dst + 3, w * v.w);
}

extern "C" void kernel_launch(void* const* d_in, const int* in_sizes, int n_in,
                              void* d_out, int out_size, void* d_ws, size_t ws_size,
                              hipStream_t stream) {
    const float* x  = (const float*)d_in[0];
    const int*   ei = (const int*)d_in[1];
    const float* W1 = (const float*)d_in[2];
    const float* b1 = (const float*)d_in[3];
    const float* W2 = (const float*)d_in[4];
    const float* b2 = (const float*)d_in[5];
    const float* W3 = (const float*)d_in[6];
    const float* b3 = (const float*)d_in[7];
    const float* g1 = (const float*)d_in[8];
    const float* g2 = (const float*)d_in[9];

    const int DIN  = 128;
    const int DH   = in_sizes[3];            // 128
    const int DOUT = in_sizes[7];            // 64
    const int N    = in_sizes[0] / DIN;      // 100000
    const int E    = in_sizes[1] / 2;        // 1600000
    const int* row = ei;
    const int* col = ei + E;
    float* out = (float*)d_out;

    // Workspace layout (floats): deg[N] | wself[N] | wedge[E] | t[N*128] | hA[N*128] | hB[N*128]
    float* ws    = (float*)d_ws;
    float* deg   = ws;
    float* wself = deg + N;
    float* wedge = wself + N;
    float* t     = wedge + E;                // byte offset multiple of 16 -> float4-safe
    float* hA    = t + (size_t)N * DH;
    float* hB    = hA + (size_t)N * DH;

    const int TB = 256;
    // --- degrees + GSO weights (reused by all 3 layers) ---
    k_zero  <<<(N + TB - 1) / TB, TB, 0, stream>>>(deg, N);
    k_degree<<<(E + TB - 1) / TB, TB, 0, stream>>>(row, deg, E);
    k_wself <<<(N + TB - 1) / TB, TB, 0, stream>>>(deg, g1, g2, wself, N);
    k_wedge <<<(E + TB - 1) / TB, TB, 0, stream>>>(deg, row, col, g1, g2, wedge, E);

    const int gemmGrid = (N + 127) / 128;    // 8 waves/block * 16 rows/wave

    // --- layer 1: t = x@W1 ; hA = wself*t + b1 ; hA[row] += wedge*t[col] ---
    k_gemm_wmma<7, false><<<gemmGrid, TB, 0, stream>>>(x, W1, t, N);
    k_init <<<((N * DH) + TB - 1) / TB, TB, 0, stream>>>(t, wself, b1, hA, N * DH, 7);
    k_scatter<<<((E * (DH / 4)) + TB - 1) / TB, TB, 0, stream>>>(t, wedge, row, col, hA,
                                                                 E * (DH / 4), 5, DH);

    // --- layer 2: t = relu(hA)@W2 ; hB = wself*t + b2 ; scatter ---
    k_gemm_wmma<7, true><<<gemmGrid, TB, 0, stream>>>(hA, W2, t, N);
    k_init <<<((N * DH) + TB - 1) / TB, TB, 0, stream>>>(t, wself, b2, hB, N * DH, 7);
    k_scatter<<<((E * (DH / 4)) + TB - 1) / TB, TB, 0, stream>>>(t, wedge, row, col, hB,
                                                                 E * (DH / 4), 5, DH);

    // --- layer 3: t = relu(hB)@W3 ; out = wself*t + b3 ; scatter into d_out ---
    k_gemm_wmma<6, true><<<gemmGrid, TB, 0, stream>>>(hB, W3, t, N);
    k_init <<<((N * DOUT) + TB - 1) / TB, TB, 0, stream>>>(t, wself, b3, out, N * DOUT, 6);
    k_scatter<<<((E * (DOUT / 4)) + TB - 1) / TB, TB, 0, stream>>>(t, wedge, row, col, out,
                                                                   E * (DOUT / 4), 4, DOUT);
}
